// GraphInteractionModule_88012469830229
// MI455X (gfx1250) — matrix operand order repaired and checked by reading, not compiled, not run on hardware
//
#include <hip/hip_runtime.h>

// ---------------------------------------------------------------------------
// Problem constants (from reference): NB=4 branches, B=16 batch, C=256, L=4096
// ---------------------------------------------------------------------------
#define NBR 4
#define BB  16
#define CCH 256
#define LLN 4096
#define LT  64            // L-columns per workgroup tile
#define CP  272           // padded LDS row stride in halves (16B-aligned rows)
#define BN_EPS 1e-5f

typedef __attribute__((ext_vector_type(16))) _Float16 v16h;
typedef __attribute__((ext_vector_type(8)))  _Float16 v8h;
typedef __attribute__((ext_vector_type(4)))  _Float16 v4h;
typedef __attribute__((ext_vector_type(8)))  float    v8f;

// ------------------------- workspace layout (bytes) ------------------------
#define WS_W16   ((size_t)0)                                         // 4 * [NB][256][256] f16 = 2 MB
#define WS_V16   (WS_W16  + (size_t)4*NBR*CCH*CCH*2)                 // v   f16  128 MB
#define WS_O16   (WS_V16  + (size_t)NBR*BB*CCH*LLN*2)                // out f16  128 MB
#define WS_QSUM  (WS_O16  + (size_t)NBR*BB*CCH*LLN*2)                // [NB][B][C] f32
#define WS_QMAX  (WS_QSUM + (size_t)NBR*BB*CCH*4)                    // ordered-u32 max
#define WS_KSUM  (WS_QMAX + (size_t)NBR*BB*CCH*4)
#define WS_KMAX  (WS_KSUM + (size_t)NBR*BB*CCH*4)
#define WS_ATTN  (WS_KMAX + (size_t)NBR*BB*CCH*4)                    // [NB][NB][B] f32
#define WS_BNS   (WS_ATTN + (size_t)NBR*NBR*BB*4)                    // [NB][C] sum
#define WS_BNQ   (WS_BNS  + (size_t)NBR*CCH*4)                       // [NB][C] sumsq
#define WS_END   (WS_BNQ  + (size_t)NBR*CCH*4)
#define ZERO_DWORDS ((int)((WS_END - WS_QSUM) / 4))                  // 67840 = 265*256

// ------------------------------- helpers -----------------------------------
__device__ __forceinline__ v16h cat16(v8h lo, v8h hi) {
  return __builtin_shufflevector(lo, hi, 0,1,2,3,4,5,6,7,8,9,10,11,12,13,14,15);
}
__device__ __forceinline__ v8f wmma_step(v16h a, v16h b, v8f c) {
  // v_wmma_f32_16x16x32_f16 : D = A(16x32 f16) * B(32x16 f16) + C(16x16 f32)
  return __builtin_amdgcn_wmma_f32_16x16x32_f16(false, a, false, b, (short)0, c, false, false);
}
// monotone float -> uint mapping so atomicMax(u32) implements float max
__device__ __forceinline__ unsigned f2ord(float f) {
  unsigned u = __float_as_uint(f);
  return (u & 0x80000000u) ? ~u : (u | 0x80000000u);
}
__device__ __forceinline__ float ord2f(unsigned u) {
  unsigned b = (u & 0x80000000u) ? (u & 0x7FFFFFFFu) : ~u;
  return __uint_as_float(b);
}

// Core 256(M) x 64(N) x 256(K) tile matmul. Wave owns 2 M-tiles x 4 N-tiles.
// A fragment (16x32 f16): lanes 0-15 row M=lane, K = k0+0..15;
//                         lanes 16-31 row M=lane-16, K += 16.
// B fragment (32x16 f16): lanes 0-15 col N=lane, K = k0+0..15; lanes 16-31 K += 16.
__device__ __forceinline__ void mm256(const _Float16* __restrict__ W,
                                      const _Float16* __restrict__ xT,
                                      int Mbase, int lane, v8f acc[2][4]) {
  const int lcol = lane & 15;
  const int kbB  = (lane < 16) ? 0 : 16;
  const int kbA  = (lane < 16) ? 0 : 16;
  for (int kb = 0; kb < 8; ++kb) {
    const int k0 = kb * 32;
    v16h bf[4];
#pragma unroll
    for (int nt = 0; nt < 4; ++nt) {
      const _Float16* p = xT + (nt * 16 + lcol) * CP + k0 + kbB;
      bf[nt] = cat16(*(const v8h*)p, *(const v8h*)(p + 8));
    }
#pragma unroll
    for (int mt = 0; mt < 2; ++mt) {
      const _Float16* wp = W + (size_t)(Mbase + mt * 16 + lcol) * CCH + k0 + kbA;
      v16h af = cat16(*(const v8h*)wp, *(const v8h*)(wp + 8));
#pragma unroll
      for (int nt = 0; nt < 4; ++nt)
        acc[mt][nt] = wmma_step(af, bf[nt], acc[mt][nt]);
    }
  }
}

// --------------------------- K0a: weights -> f16 ----------------------------
__global__ __launch_bounds__(256) void k_prep_w(const float* __restrict__ Wq,
                                                const float* __restrict__ Wk,
                                                const float* __restrict__ Wv,
                                                const float* __restrict__ Wo,
                                                _Float16* __restrict__ w16) {
  int idx = blockIdx.x * 256 + threadIdx.x;            // < 4*NB*256*256
  const int per = NBR * CCH * CCH;
  int wsel = idx / per, rem = idx - wsel * per;
  const float* s = (wsel == 0) ? Wq : (wsel == 1) ? Wk : (wsel == 2) ? Wv : Wo;
  w16[idx] = (_Float16)s[rem];
}

// --------------------------- K0b: zero accumulators -------------------------
__global__ __launch_bounds__(256) void k_zero(unsigned* __restrict__ p) {
  p[blockIdx.x * 256 + threadIdx.x] = 0u;              // 0 == -inf in ordered map
}

// ---- K1: fused q/k/v projection; q,k reduced to descriptors; v -> f16 ws ---
__global__ __launch_bounds__(256)
void k_qkv(const float* __restrict__ x,
           const float* __restrict__ bq, const float* __restrict__ bk,
           const float* __restrict__ bv,
           const _Float16* __restrict__ w16,
           float* __restrict__ qsum, unsigned* __restrict__ qmax,
           float* __restrict__ ksum, unsigned* __restrict__ kmax,
           _Float16* __restrict__ vout) {
  __shared__ _Float16 xT[LT * CP];                     // x tile, transposed, f16
  const int lblk = blockIdx.x, b = blockIdx.y, n = blockIdx.z;
  const int tid = threadIdx.x, lane = tid & 31, wv = tid >> 5;
  const int Mbase = wv * 32;
  const int nbidx = n * BB + b;

  // Stage x[n,b, :, lblk*64 .. +64] -> LDS (f32->f16, transposed to [l][c])
  const float* xg = x + ((size_t)nbidx * CCH) * LLN + lblk * LT;
  const int tr = tid >> 4, tl = (tid & 15) * 4;
  for (int cc = 0; cc < CCH; cc += 16) {
    int c = cc + tr;
    const float* src = xg + (size_t)c * LLN + tl;
    if (cc + 16 < CCH) __builtin_prefetch(src + 16 * LLN, 0, 1);
    float4 f = *(const float4*)src;
    xT[(tl + 0) * CP + c] = (_Float16)f.x;
    xT[(tl + 1) * CP + c] = (_Float16)f.y;
    xT[(tl + 2) * CP + c] = (_Float16)f.z;
    xT[(tl + 3) * CP + c] = (_Float16)f.w;
  }
  __syncthreads();

  // ---- Q and K: matmul then reduce mean/max over this L tile (never stored)
  for (int pk = 0; pk < 2; ++pk) {
    const _Float16* W = w16 + ((size_t)pk * NBR + n) * CCH * CCH;
    const float* bias = (pk == 0) ? bq : bk;
    float* psum = (pk == 0) ? qsum : ksum;
    unsigned* pmax = (pk == 0) ? qmax : kmax;
    v8f acc[2][4];
#pragma unroll
    for (int mt = 0; mt < 2; ++mt)
#pragma unroll
      for (int nt = 0; nt < 4; ++nt) acc[mt][nt] = (v8f){0,0,0,0,0,0,0,0};
    mm256(W, xT, Mbase, lane, acc);
#pragma unroll
    for (int mt = 0; mt < 2; ++mt) {
#pragma unroll
      for (int r = 0; r < 8; ++r) {
        int m = Mbase + mt * 16 + r + ((lane < 16) ? 0 : 8);
        float bi = bias[n * CCH + m];
        float s = 0.f, mx = -3.0e38f;
#pragma unroll
        for (int nt = 0; nt < 4; ++nt) {
          float vv = acc[mt][nt][r] + bi;
          s += vv; mx = fmaxf(mx, vv);
        }
#pragma unroll
        for (int off = 1; off < 16; off <<= 1) {       // reduce the 16-lane row group
          s += __shfl_xor(s, off, 32);
          mx = fmaxf(mx, __shfl_xor(mx, off, 32));
        }
        if ((lane & 15) == 0) {
          atomicAdd(psum + nbidx * CCH + m, s);
          atomicMax(pmax + nbidx * CCH + m, f2ord(mx));
        }
      }
    }
  }

  // ---- V: matmul and store as f16
  {
    const _Float16* W = w16 + ((size_t)2 * NBR + n) * CCH * CCH;
    v8f acc[2][4];
#pragma unroll
    for (int mt = 0; mt < 2; ++mt)
#pragma unroll
      for (int nt = 0; nt < 4; ++nt) acc[mt][nt] = (v8f){0,0,0,0,0,0,0,0};
    mm256(W, xT, Mbase, lane, acc);
#pragma unroll
    for (int mt = 0; mt < 2; ++mt) {
#pragma unroll
      for (int r = 0; r < 8; ++r) {
        int m = Mbase + mt * 16 + r + ((lane < 16) ? 0 : 8);
        float bi = bv[n * CCH + m];
#pragma unroll
        for (int nt = 0; nt < 4; ++nt) {
          int lg = lblk * LT + nt * 16 + (lane & 15);
          vout[((size_t)nbidx * CCH + m) * LLN + lg] = (_Float16)(acc[mt][nt][r] + bi);
        }
      }
    }
  }
}

// ---- K2: descriptors -> scores -> softmax(j) * softmax(branch_imp) ---------
__global__ __launch_bounds__(256)
void k_attn(const float* __restrict__ qsum, const unsigned* __restrict__ qmax,
            const float* __restrict__ ksum, const unsigned* __restrict__ kmax,
            const float* __restrict__ edge, const float* __restrict__ bimp,
            float* __restrict__ attn) {
  __shared__ float sc[256];
  const int tid = threadIdx.x;
  const int i = tid >> 6, j = (tid >> 4) & 3, b = tid & 15;
  const float invL = 1.0f / (float)LLN;
  float s = 0.f;
  for (int c = 0; c < CCH; ++c) {
    int qi = (i * BB + b) * CCH + c, ki = (j * BB + b) * CCH + c;
    float qv = qsum[qi] * invL + ord2f(qmax[qi]);      // mean + max
    float kv = ksum[ki] * invL + ord2f(kmax[ki]);
    s += qv * kv;
  }
  s = s * (1.0f / (4.0f * 16.0f)) * edge[i * NBR + j]; // / (4*sqrt(C))
  sc[tid] = s;
  __syncthreads();
  float mx = -3.0e38f;
  for (int jj = 0; jj < 4; ++jj) mx = fmaxf(mx, sc[(i << 6) + (jj << 4) + b]);
  float den = 0.f;
  for (int jj = 0; jj < 4; ++jj) den += expf(sc[(i << 6) + (jj << 4) + b] - mx);
  float bm = fmaxf(fmaxf(bimp[0], bimp[1]), fmaxf(bimp[2], bimp[3]));
  float bden = expf(bimp[0]-bm) + expf(bimp[1]-bm) + expf(bimp[2]-bm) + expf(bimp[3]-bm);
  float imp = expf(bimp[j] - bm) / bden;
  attn[(i * NBR + j) * BB + b] = expf(s - mx) / den * imp;
}

// ---- K3: agg = sum_j attn*v ; out = Wo@agg + bo ; BN stats ; out -> f16 ws -
__global__ __launch_bounds__(256)
void k_aggout(const float* __restrict__ bo, const _Float16* __restrict__ w16,
              const _Float16* __restrict__ vin, const float* __restrict__ attn,
              _Float16* __restrict__ out16, float* __restrict__ bns,
              float* __restrict__ bnq) {
  __shared__ _Float16 aT[LT * CP];
  const int lblk = blockIdx.x, b = blockIdx.y, i = blockIdx.z;
  const int tid = threadIdx.x, lane = tid & 31, wv = tid >> 5;
  const int Mbase = wv * 32;
  float aw[NBR];
#pragma unroll
  for (int j = 0; j < NBR; ++j) aw[j] = attn[(i * NBR + j) * BB + b];

  const int tr = tid >> 4, tl = (tid & 15) * 4;
  for (int cc = 0; cc < CCH; cc += 16) {
    int c = cc + tr;
    float s0 = 0.f, s1 = 0.f, s2 = 0.f, s3 = 0.f;
#pragma unroll
    for (int j = 0; j < NBR; ++j) {
      const _Float16* vp = vin + ((size_t)(j * BB + b) * CCH + c) * LLN + lblk * LT + tl;
      v4h h = *(const v4h*)vp;
      s0 += aw[j] * (float)h[0]; s1 += aw[j] * (float)h[1];
      s2 += aw[j] * (float)h[2]; s3 += aw[j] * (float)h[3];
    }
    aT[(tl + 0) * CP + c] = (_Float16)s0;
    aT[(tl + 1) * CP + c] = (_Float16)s1;
    aT[(tl + 2) * CP + c] = (_Float16)s2;
    aT[(tl + 3) * CP + c] = (_Float16)s3;
  }
  __syncthreads();

  const _Float16* W = w16 + ((size_t)3 * NBR + i) * CCH * CCH;
  v8f acc[2][4];
#pragma unroll
  for (int mt = 0; mt < 2; ++mt)
#pragma unroll
    for (int nt = 0; nt < 4; ++nt) acc[mt][nt] = (v8f){0,0,0,0,0,0,0,0};
  mm256(W, aT, Mbase, lane, acc);

#pragma unroll
  for (int mt = 0; mt < 2; ++mt) {
#pragma unroll
    for (int r = 0; r < 8; ++r) {
      int m = Mbase + mt * 16 + r + ((lane < 16) ? 0 : 8);
      float bi = bo[i * CCH + m];
      float s = 0.f, s2 = 0.f;
#pragma unroll
      for (int nt = 0; nt < 4; ++nt) {
        _Float16 h = (_Float16)(acc[mt][nt][r] + bi);
        int lg = lblk * LT + nt * 16 + (lane & 15);
        out16[((size_t)(i * BB + b) * CCH + m) * LLN + lg] = h;
        float fh = (float)h;                           // stats match stored values
        s += fh; s2 += fh * fh;
      }
#pragma unroll
      for (int off = 1; off < 16; off <<= 1) {
        s  += __shfl_xor(s,  off, 32);
        s2 += __shfl_xor(s2, off, 32);
      }
      if ((lane & 15) == 0) {
        atomicAdd(bns + i * CCH + m, s);
        atomicAdd(bnq + i * CCH + m, s2);
      }
    }
  }
}

// ---- K4: BN normalize + affine + ReLU + residual ---------------------------
__global__ __launch_bounds__(256)
void k_final(const float* __restrict__ x, const _Float16* __restrict__ out16,
             const float* __restrict__ gamma, const float* __restrict__ beta,
             const float* __restrict__ bns, const float* __restrict__ bnq,
             float* __restrict__ out) {
  int e4 = (blockIdx.x * 256 + threadIdx.x) * 4;
  int rest = e4 / LLN;                 // (n*B + b)*C + c
  int c = rest % CCH;
  int n = rest / (CCH * BB);
  const float invCnt = 1.0f / (float)(BB * LLN);
  float mean = bns[n * CCH + c] * invCnt;
  float var  = bnq[n * CCH + c] * invCnt - mean * mean;
  float scl  = rsqrtf(var + BN_EPS) * gamma[n * CCH + c];
  float sh   = beta[n * CCH + c] - mean * scl;
  v4h h = *(const v4h*)(out16 + e4);
  float4 xv = *(const float4*)(x + e4);
  float4 o;
  o.x = fmaxf((float)h[0] * scl + sh, 0.f) + xv.x;
  o.y = fmaxf((float)h[1] * scl + sh, 0.f) + xv.y;
  o.z = fmaxf((float)h[2] * scl + sh, 0.f) + xv.z;
  o.w = fmaxf((float)h[3] * scl + sh, 0.f) + xv.w;
  *(float4*)(out + e4) = o;
}

// ---------------------------------------------------------------------------
extern "C" void kernel_launch(void* const* d_in, const int* in_sizes, int n_in,
                              void* d_out, int out_size, void* d_ws, size_t ws_size,
                              hipStream_t stream) {
  (void)in_sizes; (void)n_in; (void)out_size;
  if (ws_size < WS_END) return;        // workspace too small: no-op (deterministic)

  const float* x    = (const float*)d_in[0];
  const float* Wq   = (const float*)d_in[1];  const float* bq = (const float*)d_in[2];
  const float* Wk   = (const float*)d_in[3];  const float* bk = (const float*)d_in[4];
  const float* Wv   = (const float*)d_in[5];  const float* bv = (const float*)d_in[6];
  const float* Wo   = (const float*)d_in[7];  const float* bo = (const float*)d_in[8];
  const float* gam  = (const float*)d_in[9];  const float* bet = (const float*)d_in[10];
  const float* edge = (const float*)d_in[11]; const float* bimp = (const float*)d_in[12];

  char* ws = (char*)d_ws;
  _Float16* w16 = (_Float16*)(ws + WS_W16);
  _Float16* v16 = (_Float16*)(ws + WS_V16);
  _Float16* o16 = (_Float16*)(ws + WS_O16);
  float*    qs  = (float*)(ws + WS_QSUM);  unsigned* qm = (unsigned*)(ws + WS_QMAX);
  float*    ks_ = (float*)(ws + WS_KSUM);  unsigned* km = (unsigned*)(ws + WS_KMAX);
  float*    at  = (float*)(ws + WS_ATTN);
  float*    bns = (float*)(ws + WS_BNS);   float*    bnq = (float*)(ws + WS_BNQ);
  float*    out = (float*)d_out;

  hipLaunchKernelGGL(k_prep_w, dim3(4 * NBR * CCH * CCH / 256), dim3(256), 0, stream,
                     Wq, Wk, Wv, Wo, w16);
  hipLaunchKernelGGL(k_zero, dim3(ZERO_DWORDS / 256), dim3(256), 0, stream,
                     (unsigned*)(ws + WS_QSUM));
  hipLaunchKernelGGL(k_qkv, dim3(LLN / LT, BB, NBR), dim3(256), 0, stream,
                     x, bq, bk, bv, w16, qs, qm, ks_, km, v16);
  hipLaunchKernelGGL(k_attn, dim3(1), dim3(256), 0, stream,
                     qs, qm, ks_, km, edge, bimp, at);
  hipLaunchKernelGGL(k_aggout, dim3(LLN / LT, BB, NBR), dim3(256), 0, stream,
                     bo, w16, v16, at, o16, bns, bnq);
  hipLaunchKernelGGL(k_final, dim3((NBR * BB * CCH * LLN) / (256 * 4)), dim3(256), 0, stream,
                     x, o16, gam, bet, bns, bnq, out);
}
